// VectorToTextConverter_65738769432652
// MI455X (gfx1250) — matrix operand order repaired and checked by reading, not compiled, not run on hardware
//
#include <hip/hip_runtime.h>

// ---------- types ----------
typedef __attribute__((ext_vector_type(16))) __bf16 v16bf;
typedef __attribute__((ext_vector_type(8)))  __bf16 v8bf;
typedef __attribute__((ext_vector_type(8)))  float  v8f;

#define BB 256
#define DD 1024
#define HH 1024
#define TT 100

__device__ inline __bf16 f2bf(float f) {
    unsigned u = __builtin_bit_cast(unsigned, f);
    unsigned r = (u + 0x7FFFu + ((u >> 16) & 1u)) >> 16;   // round-to-nearest-even
    return __builtin_bit_cast(__bf16, (unsigned short)r);
}

// ---------------------------------------------------------------------------
// WMMA GEMM:  C[M,N] = A[M,K](bf16, row-major, stride lda) * Bpacked + bias
// Bpacked layout: fragment-linear. For n-tile nt (16 cols) and k-chunk kc (32):
//   frag base = ((n/16)*(K/32) + kc)*512 elements; lane l owns 16 contiguous
//   bf16 at base + l*16, where lane l = 16*(k%32>=16) + (n%16), elem e = k%16.
// A fragment: lane half h = l>=16; elems 0..15 = K (32kc+16h..+15),
// matching Bpacked; two contiguous 16B chunks => 2x b128.
// Block = 128 threads (4 waves). Wave w: rows [by*64+16w, +16), cols [bx*64,+64).
// ---------------------------------------------------------------------------
__global__ __launch_bounds__(128) void wmma_gemm_bias(
    const __bf16* __restrict__ A, long lda,
    const __bf16* __restrict__ Bp,
    const float*  __restrict__ bias,
    float* __restrict__ C, int N, int K)
{
    const int lane = threadIdx.x & 31;
    const int wave = threadIdx.x >> 5;
    const int half = lane >> 4;
    const int l16  = lane & 15;
    const int mBase = blockIdx.y * 64 + wave * 16;
    const int nBase = blockIdx.x * 64;
    const int nKC = K >> 5;

    v8f acc[4];
#pragma unroll
    for (int nt = 0; nt < 4; ++nt) {
        float bv = bias[nBase + nt * 16 + l16];
#pragma unroll
        for (int r = 0; r < 8; ++r) acc[nt][r] = bv;
    }

    const __bf16* arow = A + (size_t)(mBase + l16) * lda + half * 16;

    for (int kc = 0; kc < nKC; ++kc) {
        v8bf a0 = *(const v8bf*)(arow);
        v8bf a1 = *(const v8bf*)(arow + 8);
        __builtin_prefetch(arow + 32, 0, 1);   // global_prefetch_b8 (gfx1250)
        arow += 32;
        v16bf af;
#pragma unroll
        for (int i = 0; i < 8; ++i) { af[i] = a0[i]; af[i + 8] = a1[i]; }

#pragma unroll
        for (int nt = 0; nt < 4; ++nt) {
            const __bf16* bp = Bp + ((size_t)((nBase >> 4) + nt) * nKC + kc) * 512
                                  + lane * 16;
            v8bf b0 = *(const v8bf*)(bp);
            v8bf b1 = *(const v8bf*)(bp + 8);
            v16bf bf;
#pragma unroll
            for (int i = 0; i < 8; ++i) { bf[i] = b0[i]; bf[i + 8] = b1[i]; }
            acc[nt] = __builtin_amdgcn_wmma_f32_16x16x32_bf16(
                false, af, false, bf, (short)0, acc[nt], false, false);
        }
    }

    // C layout: element r of v8f -> row mBase + 8*half + r, col nBase+16nt+l16
#pragma unroll
    for (int nt = 0; nt < 4; ++nt) {
        int n0 = nBase + nt * 16 + l16;
        int m0 = mBase + half * 8;
#pragma unroll
        for (int r = 0; r < 8; ++r)
            C[(size_t)(m0 + r) * N + n0] = acc[nt][r];
    }
}

// ---------------------------------------------------------------------------
// Row-wise LayerNorm over H=1024 (256 threads x 4 elems), optional ReLU.
// Writes f32 (optional) and bf16 (optional) copies.
// ---------------------------------------------------------------------------
__global__ __launch_bounds__(256) void ln_kernel(
    const float* __restrict__ in, long ldin,
    const float* __restrict__ g, const float* __restrict__ bta,
    float* __restrict__ outf, long ldof,
    __bf16* __restrict__ outb, long ldob, long oboff,
    int relu)
{
    __shared__ float s1[256], s2[256];
    const long row = blockIdx.x;
    const float* r = in + row * ldin;
    float v[4]; float a = 0.f, b = 0.f;
#pragma unroll
    for (int i = 0; i < 4; ++i) {
        v[i] = r[threadIdx.x + i * 256];
        a += v[i]; b += v[i] * v[i];
    }
    s1[threadIdx.x] = a; s2[threadIdx.x] = b;
    __syncthreads();
    for (int off = 128; off > 0; off >>= 1) {
        if ((int)threadIdx.x < off) {
            s1[threadIdx.x] += s1[threadIdx.x + off];
            s2[threadIdx.x] += s2[threadIdx.x + off];
        }
        __syncthreads();
    }
    float mean = s1[0] * (1.0f / 1024.0f);
    float var  = s2[0] * (1.0f / 1024.0f) - mean * mean;
    float inv  = rsqrtf(var + 1e-5f);
#pragma unroll
    for (int i = 0; i < 4; ++i) {
        int col = threadIdx.x + i * 256;
        float y = (v[i] - mean) * inv * g[col] + bta[col];
        if (relu) y = fmaxf(y, 0.0f);
        if (outf) outf[row * ldof + col] = y;
        if (outb) outb[row * ldob + oboff + col] = f2bf(y);
    }
}

// ---------------------------------------------------------------------------
// GRU gate fusion: r,z,n gates + state update. hn = normalized hidden (f32).
// Writes new hidden f32 (contiguous [B,H]) and bf16 at b*ldhb + hboff + h.
// ---------------------------------------------------------------------------
__global__ __launch_bounds__(256) void gru_gate(
    const float* __restrict__ gi, const float* __restrict__ gh,
    const float* __restrict__ hn,
    float* __restrict__ hf,
    __bf16* __restrict__ hb, long ldhb, long hboff)
{
    int idx = blockIdx.x * 256 + threadIdx.x;   // over B*H = 262144
    int b = idx >> 10;
    int h = idx & 1023;
    const float* gir = gi + (size_t)b * 3072;
    const float* ghr = gh + (size_t)b * 3072;
    float rg = 1.0f / (1.0f + __expf(-(gir[h]        + ghr[h])));
    float zg = 1.0f / (1.0f + __expf(-(gir[1024 + h] + ghr[1024 + h])));
    float ng = tanhf(gir[2048 + h] + rg * ghr[2048 + h]);
    float hv = hn[idx];
    float out = (1.0f - zg) * ng + zg * hv;
    hf[idx] = out;
    hb[(size_t)b * ldhb + hboff + h] = f2bf(out);
}

// ---------------------------------------------------------------------------
// Pack weight W[N,K] (f32 row-major) into bf16 WMMA B-fragment-linear layout.
// ---------------------------------------------------------------------------
__global__ __launch_bounds__(256) void pack_weight(
    const float* __restrict__ W, __bf16* __restrict__ P, int N, int K)
{
    long idx = (long)blockIdx.x * 256 + threadIdx.x;
    if (idx >= (long)N * K) return;
    int n = (int)(idx / K);
    int k = (int)(idx % K);
    int l = (((k >> 4) & 1) << 4) + (n & 15);
    int e = k & 15;
    long dst = ((long)(n >> 4) * (K >> 5) + (k >> 5)) * 512 + l * 16 + e;
    P[dst] = f2bf(W[idx]);
}

__global__ __launch_bounds__(256) void cvt_bf16(
    const float* __restrict__ in, __bf16* __restrict__ out, long n)
{
    long i = (long)blockIdx.x * 256 + threadIdx.x;
    if (i < n) out[i] = f2bf(in[i]);
}

// ---------------------------------------------------------------------------
extern "C" void kernel_launch(void* const* d_in, const int* in_sizes, int n_in,
                              void* d_out, int out_size, void* d_ws, size_t ws_size,
                              hipStream_t stream)
{
    const float* x        = (const float*)d_in[0];
    const float* enc_W1   = (const float*)d_in[1];
    const float* enc_b1   = (const float*)d_in[2];
    const float* enc_ln1g = (const float*)d_in[3];
    const float* enc_ln1b = (const float*)d_in[4];
    const float* enc_W2   = (const float*)d_in[5];
    const float* enc_b2   = (const float*)d_in[6];
    const float* enc_ln2g = (const float*)d_in[7];
    const float* enc_ln2b = (const float*)d_in[8];
    const float* Wih0     = (const float*)d_in[9];
    const float* Whh0     = (const float*)d_in[10];
    const float* bih0     = (const float*)d_in[11];
    const float* bhh0     = (const float*)d_in[12];
    const float* Wih1     = (const float*)d_in[13];
    const float* Whh1     = (const float*)d_in[14];
    const float* bih1     = (const float*)d_in[15];
    const float* bhh1     = (const float*)d_in[16];
    const float* hn_g     = (const float*)d_in[17];
    const float* hn_b     = (const float*)d_in[18];
    const float* proj_W1  = (const float*)d_in[19];
    const float* proj_b1  = (const float*)d_in[20];
    const float* proj_lng = (const float*)d_in[21];
    const float* proj_lnb = (const float*)d_in[22];
    const float* proj_W2  = (const float*)d_in[23];
    const float* proj_b2  = (const float*)d_in[24];
    float* out = (float*)d_out;

    // ---- carve workspace ----
    char* wp = (char*)d_ws;
    auto carve = [&](size_t bytes) -> char* {
        char* p = wp; wp += (bytes + 255) & ~(size_t)255; return p;
    };
    __bf16* wpE1 = (__bf16*)carve((size_t)HH * DD * 2);
    __bf16* wpE2 = (__bf16*)carve((size_t)HH * HH * 2);
    __bf16* wpI0 = (__bf16*)carve((size_t)3 * HH * HH * 2);
    __bf16* wpH0 = (__bf16*)carve((size_t)3 * HH * HH * 2);
    __bf16* wpI1 = (__bf16*)carve((size_t)3 * HH * HH * 2);
    __bf16* wpH1 = (__bf16*)carve((size_t)3 * HH * HH * 2);
    __bf16* wpP1 = (__bf16*)carve((size_t)HH * HH * 2);
    __bf16* wpP2 = (__bf16*)carve((size_t)DD * HH * 2);
    __bf16* xb   = (__bf16*)carve((size_t)BB * DD * 2);
    __bf16* e1b  = (__bf16*)carve((size_t)BB * HH * 2);
    __bf16* e2b  = (__bf16*)carve((size_t)BB * HH * 2);
    float*  c0   = (float*) carve((size_t)BB * 3 * HH * 4);
    float*  c1   = (float*) carve((size_t)BB * 3 * HH * 4);
    float*  h0f  = (float*) carve((size_t)BB * HH * 4);
    float*  h1f  = (float*) carve((size_t)BB * HH * 4);
    float*  hn0f = (float*) carve((size_t)BB * HH * 4);
    float*  hn1f = (float*) carve((size_t)BB * HH * 4);
    __bf16* hn0b = (__bf16*)carve((size_t)BB * HH * 2);
    __bf16* hn1b = (__bf16*)carve((size_t)BB * HH * 2);
    __bf16* h0b  = (__bf16*)carve((size_t)BB * HH * 2);
    __bf16* outb = (__bf16*)carve((size_t)BB * TT * HH * 2);
    float*  P1   = (float*) carve((size_t)BB * TT * HH * 4);
    __bf16* p1b  = (__bf16*)carve((size_t)BB * TT * HH * 2);
    (void)ws_size; (void)n_in; (void)in_sizes; (void)out_size;

    auto packW = [&](const float* W, __bf16* P, int N, int K) {
        long n = (long)N * K;
        pack_weight<<<dim3((unsigned)((n + 255) / 256)), 256, 0, stream>>>(W, P, N, K);
    };
    auto gemm = [&](const __bf16* A, long lda, const __bf16* Bpk,
                    const float* bias, float* C, int M, int N, int K) {
        dim3 g(N / 64, M / 64);
        wmma_gemm_bias<<<g, 128, 0, stream>>>(A, lda, Bpk, bias, C, N, K);
    };
    auto ln = [&](const float* in, long ldin, const float* g, const float* b,
                  float* of, long ldof, __bf16* ob, long ldob, long oboff,
                  int rows, int relu) {
        ln_kernel<<<dim3((unsigned)rows), 256, 0, stream>>>(
            in, ldin, g, b, of, ldof, ob, ldob, oboff, relu);
    };

    // ---- pack weights + input ----
    packW(enc_W1, wpE1, HH, DD);
    packW(enc_W2, wpE2, HH, HH);
    packW(Wih0, wpI0, 3 * HH, HH);
    packW(Whh0, wpH0, 3 * HH, HH);
    packW(Wih1, wpI1, 3 * HH, HH);
    packW(Whh1, wpH1, 3 * HH, HH);
    packW(proj_W1, wpP1, HH, HH);
    packW(proj_W2, wpP2, DD, HH);
    cvt_bf16<<<dim3((BB * DD + 255) / 256), 256, 0, stream>>>(x, xb, (long)BB * DD);

    // ---- encoder ----
    gemm(xb, DD, wpE1, enc_b1, c0, BB, HH, DD);
    ln(c0, HH, enc_ln1g, enc_ln1b, nullptr, 0, e1b, HH, 0, BB, 1);
    gemm(e1b, HH, wpE2, enc_b2, c0, BB, HH, HH);
    ln(c0, HH, enc_ln2g, enc_ln2b, nullptr, 0, e2b, HH, 0, BB, 1);

    // ---- init hidden to zero ----
    hipMemsetAsync(h0f, 0, (size_t)BB * HH * 4, stream);
    hipMemsetAsync(h1f, 0, (size_t)BB * HH * 4, stream);

    // ---- GRU recurrence (T fixed = 100) ----
    const int gateBlocks = (BB * HH) / 256;
    for (int t = 0; t < TT; ++t) {
        // hidden_norm on both layers' previous state
        ln(h0f, HH, hn_g, hn_b, hn0f, HH, hn0b, HH, 0, BB, 0);
        ln(h1f, HH, hn_g, hn_b, hn1f, HH, hn1b, HH, 0, BB, 0);

        // layer 0: input = e (t==0) else previous h1 (stored in outb)
        const __bf16* cur = (t == 0) ? e2b : (outb + (size_t)(t - 1) * HH);
        long curLda = (t == 0) ? (long)HH : (long)TT * HH;
        gemm(cur, curLda, wpI0, bih0, c0, BB, 3 * HH, HH);   // gi0
        gemm(hn0b, HH,    wpH0, bhh0, c1, BB, 3 * HH, HH);   // gh0
        gru_gate<<<dim3(gateBlocks), 256, 0, stream>>>(
            c0, c1, hn0f, h0f, h0b, HH, 0);

        // layer 1: input = new h0
        gemm(h0b, HH, wpI1, bih1, c0, BB, 3 * HH, HH);       // gi1
        gemm(hn1b, HH, wpH1, bhh1, c1, BB, 3 * HH, HH);      // gh1
        gru_gate<<<dim3(gateBlocks), 256, 0, stream>>>(
            c0, c1, hn1f, h1f, outb, (long)TT * HH, (long)t * HH);
    }

    // ---- projection over [B*T, H] ----
    gemm(outb, HH, wpP1, proj_b1, P1, BB * TT, HH, HH);
    ln(P1, HH, proj_lng, proj_lnb, nullptr, 0, p1b, HH, 0, BB * TT, 1);
    gemm(p1b, HH, wpP2, proj_b2, out, BB * TT, DD, HH);
}
